// TransformerDecoder_52604759441550
// MI455X (gfx1250) — matrix operand order repaired and checked by reading, not compiled, not run on hardware
//
#include <hip/hip_runtime.h>
#include <hip/hip_bf16.h>
#include <math.h>

// ---------------------------------------------------------------- constants
#define BB 16
#define SS 512
#define DD 768
#define HH 12
#define HD 64
#define LL 12
#define FFD 3072
#define LE 256
#define MM (BB * SS)   // 8192 token rows

typedef __attribute__((ext_vector_type(16))) __bf16 v16bf;
typedef __attribute__((ext_vector_type(8)))  __bf16 v8bf;
typedef __attribute__((ext_vector_type(8)))  float  v8f;
typedef __attribute__((ext_vector_type(4)))  float  v4f;

// ------------------------------------------------ Tensor Data Mover support
#if __has_builtin(__builtin_amdgcn_tensor_load_to_lds) && \
    __has_builtin(__builtin_amdgcn_s_wait_tensorcnt)
#define USE_TDM 1
#endif

#ifdef USE_TDM
typedef __attribute__((ext_vector_type(4))) unsigned int u32x4;
typedef __attribute__((ext_vector_type(8))) int i32x8;
typedef __attribute__((ext_vector_type(4))) int i32x4;

// DMA a 128-row x 32-elem bf16 tile (row stride = row_stride_elems) into LDS,
// padding each 64B row with 16B -> LDS stride 40 elements (= LDT).
// D# packing per CDNA5 ISA ch.8 (2D tensor, data_size=2B, type=2).
__device__ __forceinline__ void tdm_load_tile_128x32(const __bf16* gsrc,
                                                     __bf16* lds_dst,
                                                     int row_stride_elems) {
  unsigned long long ga = (unsigned long long)(uintptr_t)gsrc;
  unsigned lds = (unsigned)((unsigned long long)(uintptr_t)lds_dst);
  u32x4 g0;
  g0[0] = 1u;                                           // count=1, user desc
  g0[1] = lds;                                          // lds_addr
  g0[2] = (unsigned)ga;                                 // global_addr lo
  g0[3] = ((unsigned)(ga >> 32) & 0x01ffffffu) | (2u << 30);  // hi + type=2
  i32x8 g1;
  g1[0] = (1 << 16)    // data_size = 2 bytes
        | (1 << 20)    // pad_enable
        | (3 << 22)    // pad_interval: 16 DWORDs (64B row)
        | (3 << 25);   // pad_amount: 4 DWORDs (16B)
  g1[1] = (32 << 16);            // tensor_dim0 = 32 (lo16)
  g1[2] = (128 << 16);           // tensor_dim0 hi=0 | tensor_dim1 lo16 = 128
  g1[3] = (32 << 16);            // tensor_dim1 hi=0 | tile_dim0 = 32
  g1[4] = 128;                   // tile_dim1 = 128, tile_dim2 = 0
  g1[5] = row_stride_elems;      // tensor_dim0_stride lo32
  g1[6] = 0;                     // stride hi | dim1_stride lo
  g1[7] = 0;
  i32x4 z4 = {0, 0, 0, 0};
  i32x8 z8 = {0, 0, 0, 0, 0, 0, 0, 0};
  __builtin_amdgcn_tensor_load_to_lds(g0, g1, z4, z4, z8, 0);
}
#endif  // USE_TDM

// ---------------------------------------------------------------- WMMA utils
__device__ __forceinline__ v8f wmma_bf16(v16bf a, v16bf b, v8f c) {
  return __builtin_amdgcn_wmma_f32_16x16x32_bf16(
      false, a, false, b, (short)0, c, false, false);
}

__device__ __forceinline__ v16bf frag_from(const __bf16* p0, const __bf16* p1) {
  v8bf lo = *(const v8bf*)p0;
  v8bf hi = *(const v8bf*)p1;
  return __builtin_shufflevector(lo, hi, 0, 1, 2, 3, 4, 5, 6, 7,
                                 8, 9, 10, 11, 12, 13, 14, 15);
}

__device__ __forceinline__ v16bf load_frag_a(const __bf16* base, int ld) {
  int lane = threadIdx.x & 31;
  int row = lane & 15;
  int kb = (lane >> 4) << 3;
  const __bf16* p = base + row * ld + kb;
  return frag_from(p, p + 16);
}

__device__ __forceinline__ v16bf load_frag_bT(const __bf16* base, int ld) {
  int lane = threadIdx.x & 31;
  int col = lane & 15;
  int kb2 = (lane >> 4) << 3;
  const __bf16* p = base + col * ld + kb2;
  return frag_from(p, p + 16);
}

__device__ __forceinline__ float half_max(float v) {
#pragma unroll
  for (int m = 8; m >= 1; m >>= 1) v = fmaxf(v, __shfl_xor(v, m, 16));
  return v;
}
__device__ __forceinline__ float half_sum(float v) {
#pragma unroll
  for (int m = 8; m >= 1; m >>= 1) v += __shfl_xor(v, m, 16);
  return v;
}
__device__ __forceinline__ float gelu_f(float x) {
  return x * 0.5f * (1.0f + erff(x * 0.70710678118654752f));
}

// ---------------------------------------------------------------- small kernels
__global__ void transpose_cvt_kernel(const float* __restrict__ src,
                                     __bf16* __restrict__ dst, int R, int C) {
  __shared__ float tile[32][33];
  long zoff = (long)blockIdx.z * R * C;
  int r0 = blockIdx.y * 32, c0 = blockIdx.x * 32;
  int tx = threadIdx.x & 31, ty = threadIdx.x >> 5;
#pragma unroll
  for (int i = 0; i < 4; ++i)
    tile[ty + 8 * i][tx] = src[zoff + (long)(r0 + ty + 8 * i) * C + c0 + tx];
  __syncthreads();
#pragma unroll
  for (int i = 0; i < 4; ++i)
    dst[zoff + (long)(c0 + ty + 8 * i) * R + r0 + tx] =
        (__bf16)tile[tx][ty + 8 * i];
}

__global__ void embed_kernel(const int* __restrict__ ids,
                             const float* __restrict__ we,
                             const float* __restrict__ pe,
                             float* __restrict__ out, long total) {
  long stride = (long)gridDim.x * blockDim.x;
  for (long i = (long)blockIdx.x * blockDim.x + threadIdx.x; i < total; i += stride) {
    long r = i / DD; int c = (int)(i - r * DD);
    int s = (int)(r % SS);
    out[i] = we[(long)ids[r] * DD + c] + pe[(long)s * DD + c];
  }
}

__global__ void ln_kernel(const float* __restrict__ x,
                          const float* __restrict__ g,
                          const float* __restrict__ bta,
                          float* __restrict__ outf,
                          __bf16* __restrict__ outb, int cols) {
  int row = blockIdx.x, tid = threadIdx.x;
  const float* xr = x + (long)row * cols;
  __shared__ float sh[8];
  __shared__ float mu_s, rstd_s;

  float lsum = 0.f;
  for (int c = tid; c < cols; c += 256) lsum += xr[c];
  float v = lsum;
#pragma unroll
  for (int m = 16; m >= 1; m >>= 1) v += __shfl_xor(v, m, 32);
  if ((tid & 31) == 0) sh[tid >> 5] = v;
  __syncthreads();
  if (tid == 0) {
    float t = 0.f;
#pragma unroll
    for (int i = 0; i < 8; ++i) t += sh[i];
    mu_s = t / (float)cols;
  }
  __syncthreads();
  float mu = mu_s;

  float lv = 0.f;
  for (int c = tid; c < cols; c += 256) { float d = xr[c] - mu; lv += d * d; }
  v = lv;
#pragma unroll
  for (int m = 16; m >= 1; m >>= 1) v += __shfl_xor(v, m, 32);
  if ((tid & 31) == 0) sh[tid >> 5] = v;
  __syncthreads();
  if (tid == 0) {
    float t = 0.f;
#pragma unroll
    for (int i = 0; i < 8; ++i) t += sh[i];
    rstd_s = rsqrtf(t / (float)cols + 1e-12f);
  }
  __syncthreads();
  float rstd = rstd_s;

  for (int c = tid; c < cols; c += 256) {
    float o = (xr[c] - mu) * rstd * g[c] + bta[c];
    if (outf) outf[(long)row * cols + c] = o;
    if (outb) outb[(long)row * cols + c] = (__bf16)o;
  }
}

__global__ void ps_kernel(const int* __restrict__ epid,
                          const float* __restrict__ coefL,
                          float* __restrict__ ps) {
  int b = blockIdx.x, tid = threadIdx.x;
  __shared__ int sm[8];
  __shared__ int row_s;
  int e = epid[b * LE + tid];
#pragma unroll
  for (int m = 16; m >= 1; m >>= 1) e = max(e, __shfl_xor(e, m, 32));
  if ((tid & 31) == 0) sm[tid >> 5] = e;
  __syncthreads();
  if (tid == 0) {
    int t = 0;
#pragma unroll
    for (int i = 0; i < 8; ++i) t = max(t, sm[i]);
    int row = (t - 1) / 2;
    row_s = min(max(row, 0), 4);
  }
  __syncthreads();
  int row = row_s;
  for (int s = tid; s < SS; s += 256) {
    int idx = (s < LE) ? epid[b * LE + s] : 11;
    ps[b * SS + s] = coefL[row * 12 + idx];
  }
}

// ---------------------------------------------------------------- WMMA GEMM
#define LDT 40   // padded LDS stride (32 k + 8 pad) = 80B

__global__ void gemm_bf16_kernel(const __bf16* __restrict__ Act,  // [M,K]
                                 const __bf16* __restrict__ WT,   // [N,K]
                                 const float* __restrict__ bias,  // [N]
                                 const float* __restrict__ resid, // [M,N]|null
                                 float* __restrict__ outf,
                                 __bf16* __restrict__ outb,
                                 int Mdim, int Ndim, int Kdim, int do_gelu) {
  int tid = threadIdx.x;
  int w = tid >> 5, lane = tid & 31;
  int g = lane >> 4, cl = lane & 15;
  int m0 = blockIdx.y * 128;
  int n0 = blockIdx.x * 128;

  v8f acc[8];
#pragma unroll
  for (int mt = 0; mt < 8; ++mt)
#pragma unroll
    for (int r = 0; r < 8; ++r) acc[mt][r] = 0.f;

#ifdef USE_TDM
  // TDM double-buffered staging: wave0 drives the DMA, compute overlaps.
  __shared__ __bf16 Ws[2][128 * LDT];
  __shared__ __bf16 As[2][128 * LDT];
  int nk = Kdim / 32;
  if (w == 0) {
    tdm_load_tile_128x32(WT + (long)n0 * Kdim, &Ws[0][0], Kdim);
    tdm_load_tile_128x32(Act + (long)m0 * Kdim, &As[0][0], Kdim);
    __builtin_amdgcn_s_wait_tensorcnt(0);
  }
  __syncthreads();
  for (int ki = 0; ki < nk; ++ki) {
    int cur = ki & 1, nxt = cur ^ 1;
    if (w == 0 && ki + 1 < nk) {
      tdm_load_tile_128x32(WT + (long)n0 * Kdim + (ki + 1) * 32, &Ws[nxt][0], Kdim);
      tdm_load_tile_128x32(Act + (long)m0 * Kdim + (ki + 1) * 32, &As[nxt][0], Kdim);
    }
    v16bf af = load_frag_a(&Ws[cur][0] + (w * 16) * LDT, LDT);
    v16bf bfr[8];
#pragma unroll
    for (int mt = 0; mt < 8; ++mt)
      bfr[mt] = load_frag_bT(&As[cur][0] + (mt * 16) * LDT, LDT);
#pragma unroll
    for (int mt = 0; mt < 8; ++mt) acc[mt] = wmma_bf16(af, bfr[mt], acc[mt]);
    __syncthreads();
    if (w == 0 && ki + 1 < nk) __builtin_amdgcn_s_wait_tensorcnt(0);
    __syncthreads();
  }
#else
  __shared__ __bf16 Ws1[128 * LDT];
  __shared__ __bf16 As1[128 * LDT];
  int row = tid >> 1, seg = (tid & 1) * 16;
  for (int k0 = 0; k0 < Kdim; k0 += 32) {
    __syncthreads();
    const uint4* sw = (const uint4*)(WT + (long)(n0 + row) * Kdim + k0 + seg);
    const uint4* sa = (const uint4*)(Act + (long)(m0 + row) * Kdim + k0 + seg);
    uint4 w0 = sw[0], w1 = sw[1], a0 = sa[0], a1 = sa[1];
    uint4* dw = (uint4*)&Ws1[row * LDT + seg];
    uint4* da = (uint4*)&As1[row * LDT + seg];
    dw[0] = w0; dw[1] = w1; da[0] = a0; da[1] = a1;
    if (k0 + 32 < Kdim) {
      __builtin_prefetch(WT + (long)(n0 + row) * Kdim + k0 + 32 + seg, 0, 1);
      __builtin_prefetch(Act + (long)(m0 + row) * Kdim + k0 + 32 + seg, 0, 1);
    }
    __syncthreads();
    v16bf af = load_frag_a(Ws1 + (w * 16) * LDT, LDT);
    v16bf bfr[8];
#pragma unroll
    for (int mt = 0; mt < 8; ++mt)
      bfr[mt] = load_frag_bT(As1 + (mt * 16) * LDT, LDT);
#pragma unroll
    for (int mt = 0; mt < 8; ++mt) acc[mt] = wmma_bf16(af, bfr[mt], acc[mt]);
  }
#endif

  // epilogue: lane -> row gm, 8 consecutive n columns -> vector stores
#pragma unroll
  for (int mt = 0; mt < 8; ++mt) {
    int gm = m0 + mt * 16 + cl;
    int gn = n0 + w * 16 + 8 * g;
    float vv[8];
    v4f b0 = *(const v4f*)&bias[gn];
    v4f b1 = *(const v4f*)&bias[gn + 4];
#pragma unroll
    for (int r = 0; r < 4; ++r) vv[r] = acc[mt][r] + b0[r];
#pragma unroll
    for (int r = 0; r < 4; ++r) vv[4 + r] = acc[mt][4 + r] + b1[r];
    if (resid) {
      v4f r0 = *(const v4f*)&resid[(long)gm * Ndim + gn];
      v4f r1 = *(const v4f*)&resid[(long)gm * Ndim + gn + 4];
#pragma unroll
      for (int r = 0; r < 4; ++r) { vv[r] += r0[r]; vv[4 + r] += r1[r]; }
    }
    if (do_gelu) {
#pragma unroll
      for (int r = 0; r < 8; ++r) vv[r] = gelu_f(vv[r]);
    }
    if (outf) {
      v4f o0, o1;
#pragma unroll
      for (int r = 0; r < 4; ++r) { o0[r] = vv[r]; o1[r] = vv[4 + r]; }
      *(v4f*)&outf[(long)gm * Ndim + gn] = o0;
      *(v4f*)&outf[(long)gm * Ndim + gn + 4] = o1;
    }
    if (outb) {
      v8bf ob;
#pragma unroll
      for (int r = 0; r < 8; ++r) ob[r] = (__bf16)vv[r];
      *(v8bf*)&outb[(long)gm * Ndim + gn] = ob;
    }
  }
}

// ---------------------------------------------------------------- attention
#define AK_LD 80
#define AV_LD 40
#define AP_LD 40

__global__ void attn_kernel(const __bf16* __restrict__ qg,
                            const __bf16* __restrict__ kg,
                            const __bf16* __restrict__ vg,
                            const float* __restrict__ ps,
                            const float* __restrict__ maskv,
                            __bf16* __restrict__ ctx) {
  __shared__ __bf16 Klds[32 * AK_LD];      // [key][dim]
  __shared__ __bf16 Vt[64 * AV_LD];        // [dim][key]
  __shared__ __bf16 Plds[8 * 16 * AP_LD];  // per-wave P

  int bh = blockIdx.x;
  int b = bh / HH, h = bh % HH;
  int qchunk = blockIdx.y;
  int tid = threadIdx.x;
  int w = tid >> 5, lane = tid & 31;
  int g = lane >> 4, cl = lane & 15;
  int qbase = qchunk * 128 + w * 16;
  __bf16* pl = Plds + w * 16 * AP_LD;

  v16bf qf0, qf1;
  {
    const __bf16* qp = qg + ((long)(b * SS + qbase + cl)) * DD + h * HD;
    int kb = g << 3;
    qf0 = frag_from(qp + kb, qp + kb + 16);
    qf1 = frag_from(qp + 32 + kb, qp + 32 + kb + 16);
  }

  // per-row query mask, loaded once
  float mq[8];
#pragma unroll
  for (int r = 0; r < 8; ++r) mq[r] = maskv[b * SS + qbase + r + 8 * g];

  float Mrow[8], Srow[8], alpha[8];
#pragma unroll
  for (int r = 0; r < 8; ++r) { Mrow[r] = -1e30f; Srow[r] = 0.f; }
  v8f acc[4];
#pragma unroll
  for (int i = 0; i < 4; ++i)
#pragma unroll
    for (int r = 0; r < 8; ++r) acc[i][r] = 0.f;

  int nchunks = qchunk * 4 + 4;
  int kkey = tid >> 3, kseg = (tid & 7) * 8;
  int vdim = tid >> 2, vseg = (tid & 3) * 8;

  for (int c = 0; c < nchunks; ++c) {
    int kbase = c * 32;
    __syncthreads();
    {
      const uint4* sk =
          (const uint4*)(kg + ((long)(b * SS + kbase + kkey)) * DD + h * HD + kseg);
      *(uint4*)&Klds[kkey * AK_LD + kseg] = sk[0];
    }
    {
      v8bf vv;
#pragma unroll
      for (int j = 0; j < 8; ++j)
        vv[j] = vg[((long)(b * SS + kbase + vseg + j)) * DD + h * HD + vdim];
      *(v8bf*)&Vt[vdim * AV_LD + vseg] = vv;
    }
    __syncthreads();

    if (kbase <= qbase + 15) {
      int kb2 = g << 3;
      const __bf16* kp0 = Klds + cl * AK_LD + kb2;
      const __bf16* kp1 = Klds + (16 + cl) * AK_LD + kb2;
      v16bf kf00 = frag_from(kp0, kp0 + 16);
      v16bf kf01 = frag_from(kp0 + 32, kp0 + 48);
      v16bf kf10 = frag_from(kp1, kp1 + 16);
      v16bf kf11 = frag_from(kp1 + 32, kp1 + 48);
      v8f c0, c1;
#pragma unroll
      for (int r = 0; r < 8; ++r) { c0[r] = 0.f; c1[r] = 0.f; }
      c0 = wmma_bf16(qf0, kf00, c0);
      c1 = wmma_bf16(qf0, kf10, c1);
      c0 = wmma_bf16(qf1, kf01, c0);
      c1 = wmma_bf16(qf1, kf11, c1);

      int kk0 = kbase + cl, kk1 = kbase + 16 + cl;
      float pm0 = ps[b * SS + kk0] * 0.125f;
      float pm1 = ps[b * SS + kk1] * 0.125f;
      float km0 = maskv[b * SS + kk0];
      float km1 = maskv[b * SS + kk1];
#pragma unroll
      for (int r = 0; r < 8; ++r) {
        int qq = qbase + r + 8 * g;
        float am0 = (kk0 <= qq) ? mq[r] * km0 : 0.f;
        float am1 = (kk1 <= qq) ? mq[r] * km1 : 0.f;
        c0[r] = c0[r] * pm0 * am0 - 1e10f * (1.f - am0);
        c1[r] = c1[r] * pm1 * am1 - 1e10f * (1.f - am1);
      }
#pragma unroll
      for (int r = 0; r < 8; ++r) {
        float tmax = half_max(fmaxf(c0[r], c1[r]));
        float nM = fmaxf(Mrow[r], tmax);
        alpha[r] = __expf(Mrow[r] - nM);
        float p0 = __expf(c0[r] - nM);
        float p1 = __expf(c1[r] - nM);
        float rs = half_sum(p0 + p1);
        Srow[r] = Srow[r] * alpha[r] + rs;
        Mrow[r] = nM;
        int m = r + 8 * g;
        pl[m * AP_LD + cl] = (__bf16)p0;
        pl[m * AP_LD + 16 + cl] = (__bf16)p1;
      }
#pragma unroll
      for (int i = 0; i < 4; ++i)
#pragma unroll
        for (int r = 0; r < 8; ++r) acc[i][r] *= alpha[r];

      v16bf pa;
      {
        int kb = g << 3;
        const __bf16* pp = pl + cl * AP_LD + kb;
        pa = frag_from(pp, pp + 16);
      }
      v16bf vf[4];
#pragma unroll
      for (int i = 0; i < 4; ++i) {
        const __bf16* vp = Vt + (i * 16 + cl) * AV_LD + kb2;
        vf[i] = frag_from(vp, vp + 16);
      }
#pragma unroll
      for (int i = 0; i < 4; ++i) acc[i] = wmma_bf16(pa, vf[i], acc[i]);
    }
  }

#pragma unroll
  for (int i = 0; i < 4; ++i) {
#pragma unroll
    for (int r = 0; r < 8; ++r) {
      int m = r + 8 * g;
      int qq = qbase + m;
      int d = i * 16 + cl;
      float o = acc[i][r] / fmaxf(Srow[r], 1e-20f);
      ctx[((long)(b * SS + qq)) * DD + h * HD + d] = (__bf16)o;
    }
  }
}

// ---------------------------------------------------------------- launcher
extern "C" void kernel_launch(void* const* d_in, const int* in_sizes, int n_in,
                              void* d_out, int out_size, void* d_ws, size_t ws_size,
                              hipStream_t stream) {
  const int*   input_ids = (const int*)  d_in[0];
  const float* maskv     = (const float*)d_in[1];
  const int*   epid      = (const int*)  d_in[2];
  const float* word_emb  = (const float*)d_in[3];
  const float* pos_emb   = (const float*)d_in[4];
  const float* emb_g     = (const float*)d_in[5];
  const float* emb_b     = (const float*)d_in[6];
  const float* Wq  = (const float*)d_in[7];
  const float* bq  = (const float*)d_in[8];
  const float* Wk  = (const float*)d_in[9];
  const float* bk  = (const float*)d_in[10];
  const float* Wv  = (const float*)d_in[11];
  const float* bv  = (const float*)d_in[12];
  const float* Wo  = (const float*)d_in[13];
  const float* bo  = (const float*)d_in[14];
  const float* ln1g = (const float*)d_in[15];
  const float* ln1b = (const float*)d_in[16];
  const float* Wi  = (const float*)d_in[17];
  const float* bi  = (const float*)d_in[18];
  const float* Wo2 = (const float*)d_in[19];
  const float* bo2 = (const float*)d_in[20];
  const float* ln2g = (const float*)d_in[21];
  const float* ln2b = (const float*)d_in[22];
  const float* coef = (const float*)d_in[23];

  char* p = (char*)d_ws;
  auto alloc = [&](size_t bytes) -> void* {
    void* r = (void*)p;
    p += (bytes + 255) & ~(size_t)255;
    return r;
  };
  float*  h    = (float*) alloc((size_t)MM * DD * 4);
  float*  tmp  = (float*) alloc((size_t)MM * DD * 4);
  float*  aF   = (float*) alloc((size_t)MM * DD * 4);
  __bf16* hb   = (__bf16*)alloc((size_t)MM * DD * 2);
  __bf16* ab   = (__bf16*)alloc((size_t)MM * DD * 2);
  __bf16* qB   = (__bf16*)alloc((size_t)MM * DD * 2);
  __bf16* kB   = (__bf16*)alloc((size_t)MM * DD * 2);
  __bf16* vB   = (__bf16*)alloc((size_t)MM * DD * 2);
  __bf16* ctxB = (__bf16*)alloc((size_t)MM * DD * 2);
  __bf16* ffB  = (__bf16*)alloc((size_t)MM * FFD * 2);
  float*  psB  = (float*) alloc((size_t)BB * SS * 4);
  __bf16* wqT  = (__bf16*)alloc((size_t)LL * DD * DD * 2);
  __bf16* wkT  = (__bf16*)alloc((size_t)LL * DD * DD * 2);
  __bf16* wvT  = (__bf16*)alloc((size_t)LL * DD * DD * 2);
  __bf16* woT  = (__bf16*)alloc((size_t)LL * DD * DD * 2);
  __bf16* wiT  = (__bf16*)alloc((size_t)LL * DD * FFD * 2);
  __bf16* wo2T = (__bf16*)alloc((size_t)LL * FFD * DD * 2);

  dim3 tDD(DD / 32, DD / 32, LL);
  dim3 tDF(FFD / 32, DD / 32, LL);
  dim3 tFD(DD / 32, FFD / 32, LL);
  transpose_cvt_kernel<<<tDD, 256, 0, stream>>>(Wq,  wqT,  DD, DD);
  transpose_cvt_kernel<<<tDD, 256, 0, stream>>>(Wk,  wkT,  DD, DD);
  transpose_cvt_kernel<<<tDD, 256, 0, stream>>>(Wv,  wvT,  DD, DD);
  transpose_cvt_kernel<<<tDD, 256, 0, stream>>>(Wo,  woT,  DD, DD);
  transpose_cvt_kernel<<<tDF, 256, 0, stream>>>(Wi,  wiT,  DD, FFD);
  transpose_cvt_kernel<<<tFD, 256, 0, stream>>>(Wo2, wo2T, FFD, DD);

  embed_kernel<<<4096, 256, 0, stream>>>(input_ids, word_emb, pos_emb, tmp,
                                         (long)MM * DD);
  ln_kernel<<<MM, 256, 0, stream>>>(tmp, emb_g, emb_b, h, hb, DD);

  dim3 gD(DD / 128, MM / 128);
  dim3 gF(FFD / 128, MM / 128);
  dim3 gA(BB * HH, SS / 128);

  for (int l = 0; l < LL; ++l) {
    long oDD = (long)l * DD * DD, oDF = (long)l * DD * FFD;
    ps_kernel<<<BB, 256, 0, stream>>>(epid, coef + (long)l * 60, psB);

    gemm_bf16_kernel<<<gD, 256, 0, stream>>>(hb, wqT + oDD, bq + (long)l * DD,
                                             nullptr, nullptr, qB, MM, DD, DD, 0);
    gemm_bf16_kernel<<<gD, 256, 0, stream>>>(hb, wkT + oDD, bk + (long)l * DD,
                                             nullptr, nullptr, kB, MM, DD, DD, 0);
    gemm_bf16_kernel<<<gD, 256, 0, stream>>>(hb, wvT + oDD, bv + (long)l * DD,
                                             nullptr, nullptr, vB, MM, DD, DD, 0);

    attn_kernel<<<gA, 256, 0, stream>>>(qB, kB, vB, psB, maskv, ctxB);

    gemm_bf16_kernel<<<gD, 256, 0, stream>>>(ctxB, woT + oDD, bo + (long)l * DD,
                                             h, tmp, nullptr, MM, DD, DD, 0);
    ln_kernel<<<MM, 256, 0, stream>>>(tmp, ln1g + (long)l * DD, ln1b + (long)l * DD,
                                      aF, ab, DD);

    gemm_bf16_kernel<<<gF, 256, 0, stream>>>(ab, wiT + oDF, bi + (long)l * FFD,
                                             nullptr, nullptr, ffB, MM, FFD, DD, 1);
    gemm_bf16_kernel<<<gD, 256, 0, stream>>>(ffB, wo2T + oDF, bo2 + (long)l * DD,
                                             aF, tmp, nullptr, MM, DD, FFD, 0);
    float* outf = (l == LL - 1) ? (float*)d_out : h;
    ln_kernel<<<MM, 256, 0, stream>>>(tmp, ln2g + (long)l * DD, ln2b + (long)l * DD,
                                      outf, hb, DD);
  }
}
